// multi_head_attention_22239340659310
// MI455X (gfx1250) — matrix operand order repaired and checked by reading, not compiled, not run on hardware
//
#include <hip/hip_runtime.h>
#include <hip/hip_bf16.h>
#include <math.h>

// ---------------- problem constants ----------------
#define BB 4
#define SS 2048
#define DD 1024
#define HH 16
#define DHH 64

typedef __attribute__((ext_vector_type(16))) _Float16 v16h;
typedef __attribute__((ext_vector_type(8)))  _Float16 v8h;
typedef __attribute__((ext_vector_type(8)))  float    v8f;

__device__ __forceinline__ v8f wmma16(v16h a, v16h b, v8f c) {
  // D = A(16x32 f16) * B(32x16 f16) + C(16x16 f32)
  return __builtin_amdgcn_wmma_f32_16x16x32_f16(
      /*neg_a=*/false, a, /*neg_b=*/false, b,
      /*c_mod=*/(short)0, c, /*reuse_a=*/false, /*reuse_b=*/false);
}

__device__ __forceinline__ v8f zero8() { v8f z = {}; return z; }

// Build an A-fragment (16x32 f16) from the two 8-half groups this lane owns:
// lane<16: K {0..15}, lane>=16: K {16..31} (lh*16 folded by caller).
__device__ __forceinline__ v16h load_afrag(const _Float16* p) {
  v8h lo = *(const v8h*)p;
  v8h hi = *(const v8h*)(p + 8);
  v16h t;
#pragma unroll
  for (int c = 0; c < 8; c++) { t[c] = lo[c]; t[c + 8] = hi[c]; }
  return t;
}

// ---- CDNA5 async copy: 16B global -> LDS, tracked by ASYNCcnt ----
__device__ __forceinline__ void async_b128(unsigned lds_byte_off, const void* g) {
  asm volatile("global_load_async_to_lds_b128 %0, %1, off"
               :: "v"(lds_byte_off), "v"(g) : "memory");
}
__device__ __forceinline__ void wait_async0() {
  asm volatile("s_wait_asynccnt 0x0" ::: "memory");
}
__device__ __forceinline__ void wait_ds0() {
  asm volatile("s_wait_dscnt 0x0" ::: "memory");
}

// ---------------- fp32 -> fp16 convert (8 elems / thread) ----------------
__global__ __launch_bounds__(256) void cvt_f32_f16(const float* __restrict__ src,
                                                   _Float16* __restrict__ dst,
                                                   int n) {
  int i = (blockIdx.x * blockDim.x + threadIdx.x) * 8;
  if (i < n) {
    float4 a = *(const float4*)(src + i);
    float4 b = *(const float4*)(src + i + 4);
    v8h o;
    o[0] = (_Float16)a.x; o[1] = (_Float16)a.y;
    o[2] = (_Float16)a.z; o[3] = (_Float16)a.w;
    o[4] = (_Float16)b.x; o[5] = (_Float16)b.y;
    o[6] = (_Float16)b.z; o[7] = (_Float16)b.w;
    *(v8h*)(dst + i) = o;
  }
}

// ---------------- unified GEMM: Y = X @ W^T + b ----------------
// Xh: (B*S, D) fp16 row-major.  W: (D, D) fp16 row-major (row = out feature).
// MODE 0: outh[(b*H+h)*S*DH + s*DH + dh] = y*scale  (Q/K head layout, d = dh*16+h)
// MODE 1: outh[(b*H+h)*DH*S + dh*S + s]  = y*scale  (V transposed head layout)
// MODE 2: outf[m*D + n] = y                          (fp32 row-major, O-projection)
template <int MODE>
__global__ __launch_bounds__(256) void gemm_kernel(
    const _Float16* __restrict__ Xh, const _Float16* __restrict__ W,
    const float* __restrict__ bias, _Float16* __restrict__ outh,
    float* __restrict__ outf, float scale) {
  const int lane = threadIdx.x & 31;
  const int wave = threadIdx.x >> 5;
  const int l16 = lane & 15;
  const int lh  = lane >> 4;
  const int wm = wave & 1;   // 2 waves along M
  const int wn = wave >> 1;  // 4 waves along N
  const int waveM = blockIdx.x * 128 + wm * 64;
  const int waveN = blockIdx.y * 128 + wn * 32;

  v8f acc[4][2];
#pragma unroll
  for (int i = 0; i < 4; i++)
#pragma unroll
    for (int j = 0; j < 2; j++) acc[i][j] = zero8();

  const _Float16* pxBase[4];
  const _Float16* pwBase[2];
#pragma unroll
  for (int i = 0; i < 4; i++)
    pxBase[i] = Xh + (size_t)(waveM + i * 16 + l16) * DD + lh * 16;
#pragma unroll
  for (int j = 0; j < 2; j++)
    pwBase[j] = W + (size_t)(waveN + j * 16 + l16) * DD + lh * 16;

#pragma unroll 2
  for (int k0 = 0; k0 < DD; k0 += 32) {
    v16h a[4];
#pragma unroll
    for (int i = 0; i < 4; i++) {
      a[i] = load_afrag(pxBase[i] + k0);
      __builtin_prefetch(pxBase[i] + k0 + 64, 0, 1);  // next k-step row chunk
    }
    v16h bfr[2];
#pragma unroll
    for (int j = 0; j < 2; j++) {
      bfr[j] = *(const v16h*)(pwBase[j] + k0);
      __builtin_prefetch(pwBase[j] + k0 + 64, 0, 1);
    }
#pragma unroll
    for (int i = 0; i < 4; i++)
#pragma unroll
      for (int j = 0; j < 2; j++) acc[i][j] = wmma16(a[i], bfr[j], acc[i][j]);
  }

#pragma unroll
  for (int i = 0; i < 4; i++)
#pragma unroll
    for (int j = 0; j < 2; j++) {
      const int ncol = waveN + j * 16 + l16;
      const float bv = bias[ncol];
      const int h = ncol & (HH - 1);
      const int dh = ncol >> 4;
#pragma unroll
      for (int r = 0; r < 8; r++) {
        const int m = waveM + i * 16 + r + lh * 8;
        const float val = (acc[i][j][r] + bv) * scale;
        if (MODE == 2) {
          outf[(size_t)m * DD + ncol] = val;
        } else {
          const int b = m >> 11;  // S = 2048
          const int s = m & (SS - 1);
          size_t idx;
          if (MODE == 0)
            idx = (((size_t)(b * HH + h)) * SS + s) * DHH + dh;
          else
            idx = (((size_t)(b * HH + h)) * DHH + dh) * SS + s;
          outh[idx] = (_Float16)val;
        }
      }
    }
}

// ---------------- flash attention ----------------
// 4 waves / workgroup, each wave owns a 16-row q tile (64 q rows per WG).
// K/V blocks of 64 keys are double-buffered in LDS via async global->LDS copies
// and shared by all 4 waves.
__global__ __launch_bounds__(128) void attn_kernel(
    const _Float16* __restrict__ Qh,  // (B,H,S,DH), pre-scaled by 1/sqrt(D)
    const _Float16* __restrict__ Kh,  // (B,H,S,DH)
    const _Float16* __restrict__ Vt,  // (B,H,DH,S)
    _Float16* __restrict__ Oh) {      // (B,S,D) concat layout: col = h*64+dh
  __shared__ _Float16 sK[2][64 * 64];  // 2 x 8KB: 64 keys x 64 dh (row-major)
  __shared__ _Float16 sV[2][64 * 64];  // 2 x 8KB: 64 dh rows x 64 keys
  __shared__ _Float16 sP[4][16 * 64];  // per-wave P transpose scratch

  const int tid  = threadIdx.x;
  const int lane = tid & 31;
  const int wave = tid >> 5;
  const int l16 = lane & 15;
  const int lh  = lane >> 4;
  const int qblk = blockIdx.x & (SS / 64 - 1);  // 32 blocks of 64 q rows
  const int bh   = blockIdx.x >> 5;
  const int b = bh >> 4, h = bh & (HH - 1);
  const int qbase0 = qblk * 64;            // workgroup's first q row
  const int qbase  = qbase0 + wave * 16;   // this wave's q tile

  const _Float16* Qb = Qh + (size_t)bh * SS * DHH;
  const _Float16* Kb = Kh + (size_t)bh * SS * DHH;
  const _Float16* Vb = Vt + (size_t)bh * DHH * SS;
  _Float16* sPw = &sP[wave][0];

  // issue async copies of one 64-key K/V block into LDS buffer nb
  auto issue_block = [&](int kbase, int nb) {
#pragma unroll
    for (int it = 0; it < 4; it++) {
      const int c = tid + it * 128;  // 16B chunk id, 0..511
      // K block is contiguous: 64 rows x 128B
      async_b128((unsigned)(size_t)&sK[nb][c * 8],
                 (const void*)(Kb + (size_t)kbase * DHH + c * 8));
      // V block: 64 dh-rows x 128B, row stride S halves
      const int row = c >> 3, sub = c & 7;
      async_b128((unsigned)(size_t)&sV[nb][c * 8],
                 (const void*)(Vb + (size_t)row * SS + kbase + sub * 8));
    }
  };

  // Q A-fragments for the two 32-wide k-steps over DH=64
  v16h qf[2];
#pragma unroll
  for (int kk = 0; kk < 2; kk++)
    qf[kk] = load_afrag(Qb + (size_t)(qbase + l16) * DHH + kk * 32 + lh * 16);

  v8f oacc[4];
#pragma unroll
  for (int d = 0; d < 4; d++) oacc[d] = zero8();
  float mrow[8], lrow[8];
#pragma unroll
  for (int r = 0; r < 8; r++) { mrow[r] = -INFINITY; lrow[r] = 0.0f; }

  const int nkb = (qbase0 + 63) / 64 + 1;  // widest tile's causal range
  issue_block(0, 0);                       // prologue: stage block 0

  for (int kb = 0; kb < nkb; kb++) {
    const int kbase = kb * 64;
    const int buf = kb & 1;

    wait_async0();    // this wave's copies for block kb have landed
    __syncthreads();  // all waves' copies landed; buf^1 no longer being read
    if (kb + 1 < nkb) issue_block(kbase + 64, buf ^ 1);

    if (kbase <= qbase + 15) {  // not fully masked for this wave
      const _Float16* sKb = &sK[buf][0];
      const _Float16* sVb = &sV[buf][0];

      // S = Q K^T  (4 tiles of 16x16, K-dim 64 = 2 WMMAs each)
      v8f sacc[4];
#pragma unroll
      for (int t = 0; t < 4; t++) {
        sacc[t] = zero8();
#pragma unroll
        for (int kk = 0; kk < 2; kk++) {
          v16h bf = *(const v16h*)&sKb[(t * 16 + l16) * 64 + kk * 32 + lh * 16];
          sacc[t] = wmma16(qf[kk], bf, sacc[t]);
        }
      }

      if (kbase + 63 > qbase) {  // diagonal block: elementwise causal mask
#pragma unroll
        for (int t = 0; t < 4; t++)
#pragma unroll
          for (int r = 0; r < 8; r++) {
            const int gi = qbase + r + lh * 8;
            const int gj = kbase + t * 16 + l16;
            if (gj > gi) sacc[t][r] = -INFINITY;
          }
      }

      // online softmax: rows live at (vgpr r, lane-half lh); reduce over 16 lanes
#pragma unroll
      for (int r = 0; r < 8; r++) {
        float mp = fmaxf(fmaxf(sacc[0][r], sacc[1][r]),
                         fmaxf(sacc[2][r], sacc[3][r]));
#pragma unroll
        for (int off = 1; off < 16; off <<= 1)
          mp = fmaxf(mp, __shfl_xor(mp, off, 32));
        const float mn = fmaxf(mrow[r], mp);
        const float fac = __expf(mrow[r] - mn);
        mrow[r] = mn;
        float rs = 0.0f;
#pragma unroll
        for (int t = 0; t < 4; t++) {
          const float p = __expf(sacc[t][r] - mn);
          sacc[t][r] = p;
          rs += p;
        }
#pragma unroll
        for (int off = 1; off < 16; off <<= 1) rs += __shfl_xor(rs, off, 32);
        lrow[r] = lrow[r] * fac + rs;
#pragma unroll
        for (int d = 0; d < 4; d++) oacc[d][r] *= fac;
      }

      // transpose P (C-layout) -> A-fragment layout through per-wave LDS scratch
#pragma unroll
      for (int t = 0; t < 4; t++)
#pragma unroll
        for (int r = 0; r < 8; r++)
          sPw[(r + lh * 8) * 64 + t * 16 + l16] = (_Float16)sacc[t][r];
      wait_ds0();  // same-wave LDS RAW: stores visible before reads
      v16h pf[2];
#pragma unroll
      for (int kk = 0; kk < 2; kk++)
        pf[kk] = load_afrag(&sPw[l16 * 64 + kk * 32 + lh * 16]);

      // O += P V  (4 output d-tiles, K-dim 64 keys = 2 WMMAs each)
#pragma unroll
      for (int d = 0; d < 4; d++)
#pragma unroll
        for (int kk = 0; kk < 2; kk++) {
          v16h vf = *(const v16h*)&sVb[(d * 16 + l16) * 64 + kk * 32 + lh * 16];
          oacc[d] = wmma16(pf[kk], vf, oacc[d]);
        }
    }
  }

  // normalize + store to concat layout (B,S,D), col = h*64 + d*16 + l16
#pragma unroll
  for (int r = 0; r < 8; r++) {
    const float inv = 1.0f / lrow[r];
    const int srow = qbase + r + lh * 8;
    const size_t base = ((size_t)b * SS + srow) * DD + h * 64;
#pragma unroll
    for (int d = 0; d < 4; d++)
      Oh[base + d * 16 + l16] = (_Float16)(oacc[d][r] * inv);
  }
}

// ---------------- host side ----------------
extern "C" void kernel_launch(void* const* d_in, const int* in_sizes, int n_in,
                              void* d_out, int out_size, void* d_ws, size_t ws_size,
                              hipStream_t stream) {
  const float* q  = (const float*)d_in[0];
  const float* k  = (const float*)d_in[1];
  const float* v  = (const float*)d_in[2];
  const float* Wq = (const float*)d_in[3];
  const float* bq = (const float*)d_in[4];
  const float* Wk = (const float*)d_in[5];
  const float* bk = (const float*)d_in[6];
  const float* Wv = (const float*)d_in[7];
  const float* bv = (const float*)d_in[8];
  const float* Wo = (const float*)d_in[9];
  const float* bo = (const float*)d_in[10];
  float* out = (float*)d_out;

  _Float16* ws = (_Float16*)d_ws;
  const size_t WE = (size_t)DD * DD;       // 1M elems per weight
  const size_t XE = (size_t)BB * SS * DD;  // 8M elems per activation
  _Float16* wWq = ws;
  _Float16* wWk = wWq + WE;
  _Float16* wWv = wWk + WE;
  _Float16* wWo = wWv + WE;
  _Float16* Xq = wWo + WE;   // fp16 copies of the inputs
  _Float16* Xk = Xq + XE;
  _Float16* Xv = Xk + XE;
  _Float16* Qh = Xv + XE;    // projected, head layouts
  _Float16* Kh = Qh + XE;
  _Float16* Vt = Kh + XE;
  _Float16* Oh = Vt + XE;    // attention output, concat layout

  const int cvtB = 256;
  const int cvtGW = (int)(WE / (8 * cvtB));
  const int cvtGX = (int)(XE / (8 * cvtB));
  cvt_f32_f16<<<cvtGW, cvtB, 0, stream>>>(Wq, wWq, (int)WE);
  cvt_f32_f16<<<cvtGW, cvtB, 0, stream>>>(Wk, wWk, (int)WE);
  cvt_f32_f16<<<cvtGW, cvtB, 0, stream>>>(Wv, wWv, (int)WE);
  cvt_f32_f16<<<cvtGW, cvtB, 0, stream>>>(Wo, wWo, (int)WE);
  cvt_f32_f16<<<cvtGX, cvtB, 0, stream>>>(q, Xq, (int)XE);
  cvt_f32_f16<<<cvtGX, cvtB, 0, stream>>>(k, Xk, (int)XE);
  cvt_f32_f16<<<cvtGX, cvtB, 0, stream>>>(v, Xv, (int)XE);

  dim3 pg(BB * SS / 128, DD / 128);   // (64, 8)
  const float qscale = 1.0f / 32.0f;  // 1/sqrt(D), D = 1024
  gemm_kernel<0><<<pg, 256, 0, stream>>>(Xq, wWq, bq, Qh, nullptr, qscale);
  gemm_kernel<0><<<pg, 256, 0, stream>>>(Xk, wWk, bk, Kh, nullptr, 1.0f);
  gemm_kernel<1><<<pg, 256, 0, stream>>>(Xv, wWv, bv, Vt, nullptr, 1.0f);

  attn_kernel<<<BB * HH * (SS / 64), 128, 0, stream>>>(Qh, Kh, Vt, Oh);

  gemm_kernel<2><<<pg, 256, 0, stream>>>(Oh, wWo, bo, nullptr, out, 1.0f);
}
